// LSTMModel_48060684043002
// MI455X (gfx1250) — compile-verified, matrix operand order checked by reading, not run on hardware
//
#include <hip/hip_runtime.h>
#include <hip/hip_bf16.h>
#include <math.h>

typedef __bf16 bh_t;
typedef __attribute__((ext_vector_type(8)))  bh_t  v8bh;
typedef __attribute__((ext_vector_type(16))) bh_t  v16bh;
typedef __attribute__((ext_vector_type(8)))  float v8f;

#define HIDDEN   128
#define NGATES   512
#define TSTEPS   24
#define NSEQ     20800          // 64 * 325
#define MTILE    16
#define NTILES   1300           // NSEQ / MTILE
#define HSEQ_ELEMS ((size_t)NTILES * TSTEPS * MTILE * HIDDEN)   // bf16 elems
#define HSEQ_BYTES (HSEQ_ELEMS * 2)                             // 127,795,200
#define W0_ELEMS  (NGATES * HIDDEN)        // 65536
#define W1_ELEMS  (NGATES * 256)           // 131072

__device__ __forceinline__ float sigmf(float x) { return 1.0f / (1.0f + __expf(-x)); }

// fragment load: p points at (row base + half*8); returns {p[0..7], p[16..23]}
__device__ __forceinline__ v16bh load_frag(const bh_t* p) {
    v8bh lo = *(const v8bh*)(p);
    v8bh hi = *(const v8bh*)(p + 16);
    return __builtin_shufflevector(lo, hi, 0,1,2,3,4,5,6,7,8,9,10,11,12,13,14,15);
}

// ---------------- prep: convert weights to bf16, fuse biases ----------------
__global__ void prep_kernel(const float* __restrict__ Whh0,
                            const float* __restrict__ Wih1, const float* __restrict__ Whh1,
                            const float* __restrict__ bih0, const float* __restrict__ bhh0,
                            const float* __restrict__ bih1, const float* __restrict__ bhh1,
                            bh_t* __restrict__ w0, bh_t* __restrict__ w1,
                            float* __restrict__ bias0, float* __restrict__ bias1) {
    int idx = blockIdx.x * blockDim.x + threadIdx.x;
    int stride = gridDim.x * blockDim.x;
    for (int i = idx; i < W0_ELEMS; i += stride) w0[i] = (bh_t)Whh0[i];
    for (int i = idx; i < W1_ELEMS; i += stride) {
        int n = i >> 8, k = i & 255;
        float v = (k < HIDDEN) ? Wih1[n * HIDDEN + k] : Whh1[n * HIDDEN + (k - HIDDEN)];
        w1[i] = (bh_t)v;
    }
    for (int i = idx; i < NGATES; i += stride) {
        bias0[i] = bih0[i] + bhh0[i];
        bias1[i] = bih1[i] + bhh1[i];
    }
}

// ---------------- layer 0: K=128 recurrent WMMA + fused K=3 input proj ------
// B fragments (W_hh) are loop-invariant: hoisted into 128 VGPRs, so the hot
// loop issues only 8 ds_load_b128 per wave-step (A fragments) + 16 WMMAs.
__global__ __launch_bounds__(256) void lstm_layer0(
        const float* __restrict__ x, const float* __restrict__ Wih0,
        const bh_t* __restrict__ w0, const float* __restrict__ bias0,
        bh_t* __restrict__ hseq) {
    extern __shared__ char smem[];
    bh_t*  sW = (bh_t*)smem;                       // [512][128] bf16 = 128KB
    bh_t*  sH = (bh_t*)(smem + 131072);            // [16][128]  bf16 = 4KB
    float* sX = (float*)(smem + 131072 + 4096);    // [16][3]    f32

    const int tid  = threadIdx.x;
    const int tile = blockIdx.x;
    const int s0   = tile * MTILE;

    // cooperative weight load into LDS (8192 x uint4)
    {
        const uint4* src = (const uint4*)w0;
        uint4* dst = (uint4*)sW;
        #pragma unroll
        for (int i = 0; i < 32; ++i) dst[tid + 256 * i] = src[tid + 256 * i];
    }
    { uint4 z{0u,0u,0u,0u}; ((uint4*)sH)[tid] = z; }   // h0 = 0

    const int lane = tid & 31;
    const int w    = tid >> 5;        // wave 0..7 -> units 16w..16w+15
    const int r    = lane & 15;
    const int half = lane >> 4;
    const int j    = w * 16 + r;      // hidden unit owned by this lane

    float wih[4][3], bz[4];
    #pragma unroll
    for (int g = 0; g < 4; ++g) {
        int row = g * HIDDEN + j;
        bz[g] = bias0[row];
        #pragma unroll
        for (int f = 0; f < 3; ++f) wih[g][f] = Wih0[row * 3 + f];
    }
    float cst[8];
    #pragma unroll
    for (int m = 0; m < 8; ++m) cst[m] = 0.0f;

    __syncthreads();   // weights + h0 in LDS

    // hoist all 16 B fragments (4 gate tiles x 4 K-steps) into registers
    v16bh bfr[4][4];
    #pragma unroll
    for (int g = 0; g < 4; ++g) {
        const bh_t* bRow = sW + (g * HIDDEN + w * 16 + r) * HIDDEN + half * 8;
        #pragma unroll
        for (int k = 0; k < 4; ++k) bfr[g][k] = load_frag(bRow + k * 32);
    }

    for (int t = 0; t < TSTEPS; ++t) {
        if (tid < 48) {                       // x_t tile: [16 seq][3 feat]
            int m = tid / 3, f = tid - m * 3;
            int s = s0 + m;
            int b = s / 325, n = s - b * 325;
            sX[tid] = x[((b * 3 + f) * TSTEPS + t) * 325 + n];
        }
        // A fragments from h_{t-1}
        v16bh af[4];
        const bh_t* aRow = sH + r * HIDDEN + half * 8;
        #pragma unroll
        for (int k = 0; k < 4; ++k) af[k] = load_frag(aRow + k * 32);
        // gates = h @ W_hh^T, accumulator seeded with fused bias
        v8f acc[4];
        #pragma unroll
        for (int g = 0; g < 4; ++g) {
            float b0v = bz[g];
            v8f a = {b0v,b0v,b0v,b0v,b0v,b0v,b0v,b0v};
            #pragma unroll
            for (int k = 0; k < 4; ++k) {
                a = __builtin_amdgcn_wmma_f32_16x16x32_bf16(
                        false, af[k], false, bfr[g][k], (short)0, a, false, false);
            }
            acc[g] = a;
        }
        __syncthreads();   // sX visible, sH reads complete

        #pragma unroll
        for (int mi = 0; mi < 8; ++mi) {
            int m = half * 8 + mi;            // sequence row (C layout)
            float x0 = sX[m*3+0], x1 = sX[m*3+1], x2 = sX[m*3+2];
            float gi = acc[0][mi] + wih[0][0]*x0 + wih[0][1]*x1 + wih[0][2]*x2;
            float gf = acc[1][mi] + wih[1][0]*x0 + wih[1][1]*x1 + wih[1][2]*x2;
            float gg = acc[2][mi] + wih[2][0]*x0 + wih[2][1]*x1 + wih[2][2]*x2;
            float go = acc[3][mi] + wih[3][0]*x0 + wih[3][1]*x1 + wih[3][2]*x2;
            float c  = sigmf(gf) * cst[mi] + sigmf(gi) * tanhf(gg);
            cst[mi]  = c;
            float h  = sigmf(go) * tanhf(c);
            sH[m * HIDDEN + j] = (bh_t)h;
        }
        __syncthreads();   // h_t complete
        // coalesced spill of h_t tile to workspace (L2-resident)
        ((uint4*)(hseq + (size_t)(tile * TSTEPS + t) * (MTILE * HIDDEN)))[tid] =
            ((const uint4*)sH)[tid];
    }
}

// ---------------- layer 1: fused [x|h] K=256 WMMA + LN + FC -----------------
__global__ __launch_bounds__(256) void lstm_layer1(
        const bh_t* __restrict__ hseq, const bh_t* __restrict__ w1,
        const float* __restrict__ bias1,
        const float* __restrict__ ln_g, const float* __restrict__ ln_b,
        const float* __restrict__ fcW, const float* __restrict__ fcb,
        float* __restrict__ out) {
    extern __shared__ char smem[];
    bh_t*  sW   = (bh_t*)smem;                        // [512][256] bf16 = 256KB
    bh_t*  sA   = (bh_t*)(smem + 262144);             // [16][256]  bf16 = 8KB
    float* hfin = (float*)(smem + 262144 + 8192);     // [16][128]  f32  = 8KB

    const int tid  = threadIdx.x;
    const int tile = blockIdx.x;
    const int s0   = tile * MTILE;

    {   // weights: 16384 x uint4
        const uint4* src = (const uint4*)w1;
        uint4* dst = (uint4*)sW;
        #pragma unroll
        for (int i = 0; i < 64; ++i) dst[tid + 256 * i] = src[tid + 256 * i];
    }
    {   // zero h-part of A (cols 128..255)
        uint4 z{0u,0u,0u,0u};
        int m = tid >> 4, col = HIDDEN + (tid & 15) * 8;
        *(uint4*)(sA + m * 256 + col) = z;
    }

    const int lane = tid & 31;
    const int w    = tid >> 5;
    const int r    = lane & 15;
    const int half = lane >> 4;
    const int j    = w * 16 + r;

    float bz[4];
    #pragma unroll
    for (int g = 0; g < 4; ++g) bz[g] = bias1[g * HIDDEN + j];
    float cst[8];
    #pragma unroll
    for (int m = 0; m < 8; ++m) cst[m] = 0.0f;

    __syncthreads();

    for (int t = 0; t < TSTEPS; ++t) {
        {   // stream layer-0 h_t tile into A cols 0..127
            const uint4* src =
                (const uint4*)(hseq + (size_t)(tile * TSTEPS + t) * (MTILE * HIDDEN));
            uint4 v = src[tid];
            int m = tid >> 4, col = (tid & 15) * 8;
            *(uint4*)(sA + m * 256 + col) = v;
        }
        __syncthreads();   // A = [x_t | h_{t-1}] ready

        v16bh af[8];
        const bh_t* aRow = sA + r * 256 + half * 8;
        #pragma unroll
        for (int k = 0; k < 8; ++k) af[k] = load_frag(aRow + k * 32);

        v8f acc[4];
        #pragma unroll
        for (int g = 0; g < 4; ++g) {
            float b0v = bz[g];
            v8f a = {b0v,b0v,b0v,b0v,b0v,b0v,b0v,b0v};
            const bh_t* bRow = sW + (g * HIDDEN + w * 16 + r) * 256 + half * 8;
            #pragma unroll
            for (int k = 0; k < 8; ++k) {
                v16bh bfrag = load_frag(bRow + k * 32);
                a = __builtin_amdgcn_wmma_f32_16x16x32_bf16(
                        false, af[k], false, bfrag, (short)0, a, false, false);
            }
            acc[g] = a;
        }
        __syncthreads();   // A reads complete

        #pragma unroll
        for (int mi = 0; mi < 8; ++mi) {
            int m = half * 8 + mi;
            float gi = acc[0][mi];
            float gf = acc[1][mi];
            float gg = acc[2][mi];
            float go = acc[3][mi];
            float c  = sigmf(gf) * cst[mi] + sigmf(gi) * tanhf(gg);
            cst[mi]  = c;
            float h  = sigmf(go) * tanhf(c);
            sA[m * 256 + HIDDEN + j] = (bh_t)h;
            if (t == TSTEPS - 1) hfin[m * HIDDEN + j] = fmaxf(h, 0.0f);
        }
        __syncthreads();   // h_t (and hfin) complete
    }

    // LN + FC for this block's 16 sequences (tiny)
    if (tid < MTILE) {
        const float* hr = hfin + tid * HIDDEN;
        float mu = 0.0f;
        for (int k = 0; k < HIDDEN; ++k) mu += hr[k];
        mu *= (1.0f / HIDDEN);
        float var = 0.0f;
        for (int k = 0; k < HIDDEN; ++k) { float d = hr[k] - mu; var += d * d; }
        var *= (1.0f / HIDDEN);
        float inv = rsqrtf(var + 1e-5f);
        float o = fcb[0];
        for (int k = 0; k < HIDDEN; ++k)
            o += ((hr[k] - mu) * inv * ln_g[k] + ln_b[k]) * fcW[k];
        out[s0 + tid] = o;
    }
}

extern "C" void kernel_launch(void* const* d_in, const int* in_sizes, int n_in,
                              void* d_out, int out_size, void* d_ws, size_t ws_size,
                              hipStream_t stream) {
    const float* x    = (const float*)d_in[0];
    const float* Wih0 = (const float*)d_in[1];
    const float* Whh0 = (const float*)d_in[2];
    const float* bih0 = (const float*)d_in[3];
    const float* bhh0 = (const float*)d_in[4];
    const float* Wih1 = (const float*)d_in[5];
    const float* Whh1 = (const float*)d_in[6];
    const float* bih1 = (const float*)d_in[7];
    const float* bhh1 = (const float*)d_in[8];
    const float* lng  = (const float*)d_in[9];
    const float* lnb  = (const float*)d_in[10];
    const float* fcW  = (const float*)d_in[11];
    const float* fcb  = (const float*)d_in[12];

    char* ws = (char*)d_ws;
    bh_t*  hseq = (bh_t*)ws;
    bh_t*  w0   = (bh_t*)(ws + HSEQ_BYTES);
    bh_t*  w1   = (bh_t*)(ws + HSEQ_BYTES + (size_t)W0_ELEMS * 2);
    float* b0   = (float*)(ws + HSEQ_BYTES + (size_t)W0_ELEMS * 2 + (size_t)W1_ELEMS * 2);
    float* b1   = b0 + NGATES;

    prep_kernel<<<256, 256, 0, stream>>>(Whh0, Wih1, Whh1, bih0, bhh0, bih1, bhh1,
                                         w0, w1, b0, b1);

    size_t smem0 = 131072 + 4096 + 16 * 3 * sizeof(float);         // ~135.4 KB
    lstm_layer0<<<NTILES, 256, smem0, stream>>>(x, Wih0, w0, b0, hseq);

    size_t smem1 = 262144 + 8192 + 8192;                           // 272 KB
    lstm_layer1<<<NTILES, 256, smem1, stream>>>(hseq, w1, b1, lng, lnb, fcW, fcb,
                                                (float*)d_out);
}
